// MyNet_45174466019733
// MI455X (gfx1250) — matrix operand order repaired and checked by reading, not compiled, not run on hardware
//
#include <hip/hip_runtime.h>
#include <hip/hip_bf16.h>

// Problem constants (from reference)
#define N_   256
#define T_   128
#define V_   128
#define U_   512
#define NB_  8
#define DIM_ 128
#define NC_  4
#define KREC 640            // U + V
#define KC   20             // KREC/32 k-chunks for the scan
#define KC1  128            // 4096/32 k-chunks for readout
#define ROWS 32             // batch rows per scan WG (2 M-tiles)

typedef __attribute__((ext_vector_type(16))) __bf16 v16bf;
typedef __attribute__((ext_vector_type(8)))  __bf16 v8bf;
typedef __attribute__((ext_vector_type(8)))  float  v8f;

// A-fragment (16x32 bf16) LDS placement: given row m (0..15) and k-in-chunk kl (0..31),
// return lane and half-index per the CDNA5 16-bit A layout:
// lanes 0-15 hold K 0..7 (hv 0..7) and 16..23 (hv 8..15); lanes 16-31 hold 8..15 and 24..31.
__device__ __forceinline__ void afrag(int kl, int m, int* l, int* hv) {
    *l  = m + ((kl >> 3) & 1 ? 16 : 0);
    *hv = (kl & 7) + ((kl & 16) ? 8 : 0);
}

// ---------------------------------------------------------------------------
// Pack [W_int | W_in] (fp32) -> bf16 B-fragments: wcomb[b][c(20)][j(32)][lane(32)][hv(16)]
// B-frag element (lane l, half hv) = Bmat[k][u],  k = c*32 + 8*(l>>4) + (hv&7) + 16*(hv>>3),  u = j*16 + (l&15)
// Bmat[k][u] = k<512 ? W_int[b][u][k] : W_in[b][u][k-512]
// ---------------------------------------------------------------------------
__global__ void pack_wcomb_kernel(const float* __restrict__ Wint,
                                  const float* __restrict__ Win,
                                  __bf16* __restrict__ wcomb) {
    long i = (long)blockIdx.x * blockDim.x + threadIdx.x;
    const long total = (long)NB_ * KC * 32 * 32 * 16;
    if (i >= total) return;
    int hv = i & 15;
    int l  = (i >> 4) & 31;
    int j  = (i >> 9) & 31;
    long rest = i >> 14;
    int c = (int)(rest % KC);
    int b = (int)(rest / KC);
    int k = c * 32 + ((l & 16) ? 8 : 0) + (hv & 7) + ((hv & 8) ? 16 : 0);
    int u = j * 16 + (l & 15);
    float v = (k < U_) ? Wint[((long)b * U_ + u) * U_ + k]
                       : Win[((long)b * U_ + u) * V_ + (k - U_)];
    wcomb[i] = (__bf16)v;
}

// ---------------------------------------------------------------------------
// Pack W1 (T,4096,128 fp32) -> bf16 B-fragments: w1p[t][c(128)][j(8)][lane(32)][hv(16)]
// ---------------------------------------------------------------------------
__global__ void pack_w1_kernel(const float* __restrict__ W1,
                               __bf16* __restrict__ w1p) {
    long i = (long)blockIdx.x * blockDim.x + threadIdx.x;
    const long total = (long)T_ * KC1 * 8 * 32 * 16;   // 2^26
    if (i >= total) return;
    int hv = i & 15;
    int l  = (i >> 4) & 31;
    int j  = (i >> 9) & 7;
    int c  = (i >> 12) & 127;
    int t  = (int)(i >> 19);
    int k  = c * 32 + ((l & 16) ? 8 : 0) + (hv & 7) + ((hv & 8) ? 16 : 0);
    int d  = j * 16 + (l & 15);
    w1p[i] = (__bf16)W1[((long)t * (NB_ * U_) + k) * DIM_ + d];
}

// ---------------------------------------------------------------------------
// Recurrent scan. One WG = (block b, 32 batch rows), 256 threads = 8 waves.
// Per step: fused GEMM [32 x 640] x [640 x 512] via v_wmma_f32_16x16x32_bf16,
// tanh, write h' back to LDS (A-frag layout) and to global states (bf16).
// Weights are re-streamed from L2 each step; an opaque byte-offset (asm) keeps
// LLVM from hoisting (and spilling) the fragment loads across t while
// preserving the global address space of the loads (global_load, not flat).
// ---------------------------------------------------------------------------
__global__ void __launch_bounds__(256)
scan_kernel(const float* __restrict__ X,
            const __bf16* __restrict__ wcomb,
            __bf16* __restrict__ states) {
    __shared__ __align__(32) __bf16 hxA[2][KC][32][16];  // 40 KB: two M-tiles of [h | x_t]

    const int b  = blockIdx.x >> 3;                 // reservoir block (8)
    const int n0 = (blockIdx.x & 7) * ROWS;         // batch-row tile (8 per block)
    const int tid  = threadIdx.x;
    const int lane = tid & 31;
    const int wave = tid >> 5;                      // 0..7, each owns 4 N-tiles

    // zero LDS state (h starts at 0; x chunks overwritten every step)
    for (int i = tid; i < 2 * KC * 32 * 16; i += 256)
        ((__bf16*)hxA)[i] = (__bf16)0.0f;

    const __bf16* wb = wcomb + (long)b * KC * 32 * 32 * 16;

    for (int t = 0; t < T_; ++t) {
        // opaque zero offset: blocks LICM across the t-loop but keeps the
        // pointer's provenance (so loads stay global_load_b128, not flat)
        long woff = 0;
        asm volatile("" : "+s"(woff));
        const __bf16* wbt = wb + woff;

        __syncthreads();
        // stage x_t (32 rows x 128 fp32) into A-frag chunks 16..19
        for (int i = tid; i < ROWS * V_; i += 256) {
            int m = i >> 7, v = i & 127;
            float xv = X[((long)(n0 + m) * T_ + t) * V_ + v];
            int k = U_ + v;
            int l2, hv2; afrag(k & 31, m & 15, &l2, &hv2);
            hxA[m >> 4][k >> 5][l2][hv2] = (__bf16)xv;
        }
        __syncthreads();

        v8f acc[2][4] = {};
        for (int c = 0; c < KC; ++c) {
            v16bf a0 = *(const v16bf*)&hxA[0][c][lane][0];
            v16bf a1 = *(const v16bf*)&hxA[1][c][lane][0];
            if (c + 1 < KC)   // warm near caches for next weight chunk
                __builtin_prefetch(wbt + ((((long)(c + 1) * 32 + wave * 4) * 32) + lane) * 16, 0, 3);
#pragma unroll
            for (int j4 = 0; j4 < 4; ++j4) {
                int j = wave * 4 + j4;
                v16bf bf = *(const v16bf*)(wbt + (((long)c * 32 + j) * 32 + lane) * 16);
                acc[0][j4] = __builtin_amdgcn_wmma_f32_16x16x32_bf16(
                    false, a0, false, bf, (short)0, acc[0][j4], false, false);
                acc[1][j4] = __builtin_amdgcn_wmma_f32_16x16x32_bf16(
                    false, a1, false, bf, (short)0, acc[1][j4], false, false);
            }
        }
        __syncthreads();   // all waves done reading h before overwrite

        // h' = tanh(acc): D-tile (lane,r) -> row m, col u; write LDS A-frag + global states
#pragma unroll
        for (int mt = 0; mt < 2; ++mt) {
#pragma unroll
            for (int j4 = 0; j4 < 4; ++j4) {
                int j = wave * 4 + j4;
                int u = j * 16 + (lane & 15);
                int mbase = ((lane & 16) ? 8 : 0);
#pragma unroll
                for (int r = 0; r < 8; ++r) {
                    float  hv = tanhf(acc[mt][j4][r]);
                    __bf16 hb = (__bf16)hv;
                    int l2, h2; afrag(u & 31, mbase + r, &l2, &h2);
                    hxA[mt][u >> 5][l2][h2] = hb;
                    states[(((long)t * NB_ + b) * N_ + (n0 + mt * 16 + mbase + r)) * U_ + u] = hb;
                }
            }
        }
    }
}

// ---------------------------------------------------------------------------
// Readout: per (t, 16-row tile): hid = tanh(feat @ W1[t] + b1[t]); pred = hid @ W2[t] + b2[t]
// 128 threads = 4 waves, each wave owns 2 of 8 DIM-tiles; K = 4096 (128 chunks).
// ---------------------------------------------------------------------------
__global__ void __launch_bounds__(128)
readout_kernel(const __bf16* __restrict__ states,
               const __bf16* __restrict__ w1p,
               const float* __restrict__ b1,
               const float* __restrict__ W2,
               const float* __restrict__ b2,
               float* __restrict__ out) {
    __shared__ float hid[16][DIM_];

    const int t  = blockIdx.x >> 4;
    const int n0 = (blockIdx.x & 15) * 16;
    const int tid  = threadIdx.x;
    const int lane = tid & 31;
    const int wave = tid >> 5;               // 0..3

    const __bf16* w1t = w1p + (long)t * KC1 * 8 * 32 * 16;

    v8f acc[2] = {};
    for (int c = 0; c < KC1; ++c) {
        // A-frag from states: lane holds row n0+(lane&15), k runs [k1..k1+7] and [k1+16..k1+23]
        int k1 = c * 32 + ((lane & 16) ? 8 : 0);
        int bblk = k1 >> 9, u1 = k1 & 511;
        const __bf16* sp = states + (((long)t * NB_ + bblk) * N_ + (n0 + (lane & 15))) * U_;
        v8bf alo = *(const v8bf*)(sp + u1);
        v8bf ahi = *(const v8bf*)(sp + u1 + 16);
        v16bf a = __builtin_shufflevector(alo, ahi,
                    0,1,2,3,4,5,6,7, 8,9,10,11,12,13,14,15);
#pragma unroll
        for (int jj = 0; jj < 2; ++jj) {
            int j = wave * 2 + jj;
            v16bf bf = *(const v16bf*)(w1t + (((long)c * 8 + j) * 32 + lane) * 16);
            acc[jj] = __builtin_amdgcn_wmma_f32_16x16x32_bf16(
                false, a, false, bf, (short)0, acc[jj], false, false);
        }
    }

    // hid = tanh(C + b1) into LDS
#pragma unroll
    for (int jj = 0; jj < 2; ++jj) {
        int d = (wave * 2 + jj) * 16 + (lane & 15);
        int m = ((lane & 16) ? 8 : 0);
        float bias = b1[(long)t * DIM_ + d];
#pragma unroll
        for (int r = 0; r < 8; ++r)
            hid[m + r][d] = tanhf(acc[jj][r] + bias);
    }
    __syncthreads();

    // tiny head: 16 rows x 4 classes, dot over 128
    if (tid < 64) {
        int m = tid >> 2, cc = tid & 3;
        float s = b2[(long)t * NC_ + cc];
        const float* w2t = W2 + (long)t * DIM_ * NC_;
        for (int d = 0; d < DIM_; ++d)
            s += hid[m][d] * w2t[d * NC_ + cc];
        out[((long)t * N_ + n0 + m) * NC_ + cc] = s;
    }
}

// ---------------------------------------------------------------------------
extern "C" void kernel_launch(void* const* d_in, const int* in_sizes, int n_in,
                              void* d_out, int out_size, void* d_ws, size_t ws_size,
                              hipStream_t stream) {
    const float* X    = (const float*)d_in[0];   // (N,T,V)
    const float* Wint = (const float*)d_in[1];   // (NB,U,U)
    const float* Win  = (const float*)d_in[2];   // (NB,U,V)
    const float* W1   = (const float*)d_in[3];   // (T,NB*U,DIM)
    const float* b1   = (const float*)d_in[4];   // (T,DIM)
    const float* W2   = (const float*)d_in[5];   // (T,DIM,NC)
    const float* b2   = (const float*)d_in[6];   // (T,NC)
    float* out = (float*)d_out;

    // workspace layout (bytes)
    char* ws = (char*)d_ws;
    __bf16* states = (__bf16*)ws;                                    // 268,435,456 B
    __bf16* wcomb  = (__bf16*)(ws + (size_t)268435456);              //   5,242,880 B
    __bf16* w1p    = (__bf16*)(ws + (size_t)268435456 + 5242880);    // 134,217,728 B

    {   // pack recurrent + input weights into bf16 fragments
        long total = (long)NB_ * KC * 32 * 32 * 16;
        pack_wcomb_kernel<<<(unsigned)((total + 255) / 256), 256, 0, stream>>>(Wint, Win, wcomb);
    }
    {   // pack readout W1 into bf16 fragments
        long total = (long)T_ * KC1 * 8 * 32 * 16;
        pack_w1_kernel<<<(unsigned)((total + 255) / 256), 256, 0, stream>>>(W1, w1p);
    }
    // serial-in-T reservoir scan, parallel over NB x (N/32) = 64 WGs
    scan_kernel<<<NB_ * (N_ / ROWS), 256, 0, stream>>>(X, wcomb, states);
    // per-timestep MLP readout, T x (N/16) = 2048 WGs
    readout_kernel<<<T_ * (N_ / 16), 128, 0, stream>>>(states, w1p, b1, W2, b2, out);
}
